// _ProbabilisticEncoder_38903813767467
// MI455X (gfx1250) — matrix operand order repaired and checked by reading, not compiled, not run on hardware
//
#include <hip/hip_runtime.h>
#include <stdint.h>

#define D_MODEL 1024
#define VOCAB   50257
#define VP      50432      // 197 * 256, padded vocab
#define NBLK    197        // VP / BN
#define ROWS    4096       // 2 * 2048 tokens
#define BM      32
#define BN      256
#define KC      32

typedef __attribute__((ext_vector_type(16))) __bf16 v16bf;
typedef __attribute__((ext_vector_type(8)))  float  v8f;

union Frag { v16bf v; unsigned int u[8]; };

// fp32 pair -> packed bf16 via round-half-up + v_perm_b32 (3 VALU total)
__device__ __forceinline__ unsigned int pack2(float a, float b) {
    unsigned int ua = __float_as_uint(a) + 0x8000u;
    unsigned int ub = __float_as_uint(b) + 0x8000u;
    return __builtin_amdgcn_perm(ub, ua, 0x07060302u); // {ub[31:16], ua[31:16]}
}
__device__ __forceinline__ float bflo(unsigned int u) { return __uint_as_float((u & 0xFFFFu) << 16); }
__device__ __forceinline__ float bfhi(unsigned int u) { return __uint_as_float(u & 0xFFFF0000u); }

__device__ __forceinline__ void sched_hint_ds_then_wmma() {
#if __has_builtin(__builtin_amdgcn_sched_group_barrier)
    __builtin_amdgcn_sched_group_barrier(0x100, 10, 0); // 10 DS reads first
    __builtin_amdgcn_sched_group_barrier(0x008, 4, 0);  // then 4 WMMA back-to-back
#endif
}

// ---------------------------------------------------------------- GEMM1
// logits[32x256 tile] = X (bf16) * W^T (bf16) + bias, store bf16 logits +
// per-(row, n-block) online softmax partials (max, sumexp).
// Double-buffered LDS; global loads for tile k+1 issued before WMMAs of tile k.
__global__ __launch_bounds__(256) void k_gemm1(const float* __restrict__ X,
        const float* __restrict__ W, const float* __restrict__ bias,
        unsigned short* __restrict__ logits, float2* __restrict__ part)
{
    __shared__ unsigned int sX[2][BM][KC / 2];   // packed bf16 pairs (k, k+1)
    __shared__ unsigned int sW[2][BN][KC / 2];
    __shared__ float        sLog[BM][BN + 4];    // pad to dodge bank conflicts

    const int tid   = threadIdx.x;
    const int lane  = tid & 31;
    const int wid   = tid >> 5;
    const int wm    = wid >> 2;                  // 0..1  (M sub-tile)
    const int wn    = wid & 3;                   // 0..3  (N sub-tile, 64 cols each)
    const int khalf = lane >> 4;
    const int n0    = blockIdx.x * BN;
    const int m0    = blockIdx.y * BM;
    const bool fullN = (n0 + BN) <= VOCAB;       // uniform fast-path predicate

    const v8f vz = {0.f,0.f,0.f,0.f,0.f,0.f,0.f,0.f};
    v8f acc[4] = {vz, vz, vz, vz};

    const int srow = tid >> 3;                   // 0..31
    const int skf  = (tid & 7) * 4;              // float k offset within KC

    float xf[4];
    float wf[8][4];

    auto ldregs = [&](int kk) {
        const float* sx = X + (size_t)(m0 + srow) * D_MODEL + kk + skf;
        xf[0] = sx[0]; xf[1] = sx[1]; xf[2] = sx[2]; xf[3] = sx[3];
        if (fullN) {                             // uniform branch: no lane guards
            #pragma unroll
            for (int p = 0; p < 8; ++p) {
                const float* sw = W + (size_t)(n0 + p * 32 + srow) * D_MODEL + kk + skf;
                wf[p][0] = sw[0]; wf[p][1] = sw[1]; wf[p][2] = sw[2]; wf[p][3] = sw[3];
                if (kk + 2 * KC < D_MODEL) __builtin_prefetch(sw + 2 * KC, 0, 1);
            }
        } else {
            #pragma unroll
            for (int p = 0; p < 8; ++p) {
                int n = n0 + p * 32 + srow;
                wf[p][0] = wf[p][1] = wf[p][2] = wf[p][3] = 0.f;
                if (n < VOCAB) {
                    const float* sw = W + (size_t)n * D_MODEL + kk + skf;
                    wf[p][0] = sw[0]; wf[p][1] = sw[1]; wf[p][2] = sw[2]; wf[p][3] = sw[3];
                }
            }
        }
    };
    auto stregs = [&](int b) {
        sX[b][srow][skf >> 1]       = pack2(xf[0], xf[1]);
        sX[b][srow][(skf >> 1) + 1] = pack2(xf[2], xf[3]);
        #pragma unroll
        for (int p = 0; p < 8; ++p) {
            sW[b][p * 32 + srow][skf >> 1]       = pack2(wf[p][0], wf[p][1]);
            sW[b][p * 32 + srow][(skf >> 1) + 1] = pack2(wf[p][2], wf[p][3]);
        }
    };

    ldregs(0);
    stregs(0);
    __syncthreads();

    int buf = 0;
    for (int kk = 0; kk < D_MODEL; kk += KC) {
        const bool more = (kk + KC) < D_MODEL;
        if (more) ldregs(kk + KC);               // global stream for next tile

        Frag A;
        const int ar = wm * 16 + (lane & 15);
        #pragma unroll
        for (int v = 0; v < 8; ++v) {            // A layout: k = khalf*16 + 2v
            int kp = khalf * 8 + v;
            A.u[v] = sX[buf][ar][kp];
        }
        const int bc = wn * 64 + (lane & 15);
        Frag B[4];
        #pragma unroll
        for (int f = 0; f < 4; ++f)
            #pragma unroll
            for (int v = 0; v < 8; ++v)          // B layout: k = khalf*16 + 2v
                B[f].u[v] = sW[buf][bc + f * 16][khalf * 8 + v];
        #pragma unroll
        for (int f = 0; f < 4; ++f)
            acc[f] = __builtin_amdgcn_wmma_f32_16x16x32_bf16(
                false, A.v, false, B[f].v, (short)0, acc[f], false, false);
        sched_hint_ds_then_wmma();

        if (more) stregs(buf ^ 1);
        __syncthreads();
        buf ^= 1;
    }

    // bias add + spill tile to LDS (OOB vocab columns -> -1e30 so exp()==0)
    const int crow = wm * 16 + khalf * 8;
    #pragma unroll
    for (int f = 0; f < 4; ++f) {
        int  ncol = wn * 64 + f * 16 + (lane & 15);
        int  nG   = n0 + ncol;
        bool ok   = nG < VOCAB;
        float bv  = ok ? bias[nG] : 0.f;
        #pragma unroll
        for (int r = 0; r < 8; ++r)
            sLog[crow + r][ncol] = ok ? (acc[f][r] + bv) : -1e30f;
    }
    __syncthreads();
    // per-row partial max & sumexp over this 256-col block (1 wave per row)
    #pragma unroll
    for (int pass = 0; pass < 4; ++pass) {
        int row = pass * 8 + wid;
        float mx = -3e38f;
        #pragma unroll
        for (int j = 0; j < 8; ++j) mx = fmaxf(mx, sLog[row][lane * 8 + j]);
        #pragma unroll
        for (int off = 16; off; off >>= 1) mx = fmaxf(mx, __shfl_xor(mx, off, 32));
        float sm = 0.f;
        #pragma unroll
        for (int j = 0; j < 8; ++j) sm += __expf(sLog[row][lane * 8 + j] - mx);
        #pragma unroll
        for (int off = 16; off; off >>= 1) sm += __shfl_xor(sm, off, 32);
        if (lane == 0) part[(size_t)(m0 + row) * NBLK + blockIdx.x] = make_float2(mx, sm);
    }
    {   // coalesced bf16 logits store
        int row = tid >> 3;
        int c0  = (tid & 7) * 32;
        unsigned int* dst = (unsigned int*)(logits + (size_t)(m0 + row) * VP + n0 + c0);
        #pragma unroll
        for (int j = 0; j < 16; ++j)
            dst[j] = pack2(sLog[row][c0 + 2 * j], sLog[row][c0 + 2 * j + 1]);
    }
}

// ---------------------------------------------------------------- softmax reduce
__device__ __forceinline__ void merge2(float& m, float& l, float m2, float l2) {
    float M = fmaxf(m, m2);
    l = l * __expf(m - M) + l2 * __expf(m2 - M);
    m = M;
}

__global__ __launch_bounds__(256) void k_softmax_reduce(const float2* __restrict__ part,
        float* __restrict__ rowM, float* __restrict__ rowInv)
{
    int row = blockIdx.x, t = threadIdx.x;
    float m = -3e38f, l = 0.f;
    for (int j = t; j < NBLK; j += 256) {
        float2 p = part[(size_t)row * NBLK + j];
        merge2(m, l, p.x, p.y);
    }
    __shared__ float sm[256], sl[256];
    sm[t] = m; sl[t] = l;
    __syncthreads();
    for (int s = 128; s > 0; s >>= 1) {
        if (t < s) {
            float mm = sm[t], ll = sl[t];
            merge2(mm, ll, sm[t + s], sl[t + s]);
            sm[t] = mm; sl[t] = ll;
        }
        __syncthreads();
    }
    if (t == 0) { rowM[row] = sm[0]; rowInv[row] = 1.0f / sl[0]; }
}

// ---------------------------------------------------------------- logits -> probs (in place, bf16 pairs)
__global__ __launch_bounds__(256) void k_softmaxify(unsigned int* __restrict__ probs,
        const float* __restrict__ rowM, const float* __restrict__ rowInv)
{
    int row = blockIdx.y;
    int idx = blockIdx.x * 256 + threadIdx.x;
    if (idx >= VP / 2) return;
    unsigned int* p = probs + (size_t)row * (VP / 2) + idx;
    unsigned int v  = *p;
    float m = rowM[row], inv = rowInv[row];
    *p = pack2(__expf(bflo(v) - m) * inv, __expf(bfhi(v) - m) * inv);
}

// ---------------------------------------------------------------- GEMM2: soft = P * E
__global__ __launch_bounds__(256) void k_gemm2(const unsigned int* __restrict__ probs,
        const float* __restrict__ emb, float* __restrict__ soft)
{
    __shared__ unsigned int sA[2][BM][KC / 2];
    __shared__ unsigned int sB[2][BN][KC / 2];

    const int tid   = threadIdx.x;
    const int lane  = tid & 31;
    const int wid   = tid >> 5;
    const int wm    = wid >> 2;
    const int wn    = wid & 3;
    const int khalf = lane >> 4;
    const int n0    = blockIdx.x * BN;           // over D_MODEL (4 blocks)
    const int m0    = blockIdx.y * BM;

    const v8f vz = {0.f,0.f,0.f,0.f,0.f,0.f,0.f,0.f};
    v8f acc[4] = {vz, vz, vz, vz};

    const int arow = tid >> 3;                   // A staging: 0..31
    const int akp  = (tid & 7) * 2;
    const int bkp  = tid >> 4;                   // B staging: k-pair 0..15
    const int bnb  = (tid & 15) * 16;            // 16 cols per thread

    unsigned int af[2];
    float b0f[16], b1f[16];

    auto ldregs = [&](int kk) {
        const unsigned int* sp = probs + ((size_t)(m0 + arow) * VP + kk) / 2 + akp;
        af[0] = sp[0]; af[1] = sp[1];
        __builtin_prefetch(sp + KC / 2, 0, 1);
        const int k0 = kk + 2 * bkp, k1 = k0 + 1;
        const float* r0 = emb + (size_t)k0 * D_MODEL + n0 + bnb;
        const float* r1 = emb + (size_t)k1 * D_MODEL + n0 + bnb;
        if (kk + KC <= VOCAB) {                  // uniform fast path
            #pragma unroll
            for (int i = 0; i < 16; ++i) { b0f[i] = r0[i]; b1f[i] = r1[i]; }
            __builtin_prefetch(r0 + KC * D_MODEL, 0, 1);
        } else {
            bool v0 = k0 < VOCAB, v1 = k1 < VOCAB;
            #pragma unroll
            for (int i = 0; i < 16; ++i) {
                b0f[i] = v0 ? r0[i] : 0.f;
                b1f[i] = v1 ? r1[i] : 0.f;
            }
        }
    };
    auto stregs = [&](int b) {
        sA[b][arow][akp]     = af[0];
        sA[b][arow][akp + 1] = af[1];
        #pragma unroll
        for (int i = 0; i < 16; ++i)
            sB[b][bnb + i][bkp] = pack2(b0f[i], b1f[i]);
    };

    ldregs(0);
    stregs(0);
    __syncthreads();

    int buf = 0;
    for (int kk = 0; kk < VP; kk += KC) {
        const bool more = (kk + KC) < VP;
        if (more) ldregs(kk + KC);

        Frag A;
        const int ar = wm * 16 + (lane & 15);
        #pragma unroll
        for (int v = 0; v < 8; ++v) {
            int kp = khalf * 8 + v;
            A.u[v] = sA[buf][ar][kp];
        }
        const int bc = wn * 64 + (lane & 15);
        Frag B[4];
        #pragma unroll
        for (int f = 0; f < 4; ++f)
            #pragma unroll
            for (int v = 0; v < 8; ++v)
                B[f].u[v] = sB[buf][bc + f * 16][khalf * 8 + v];
        #pragma unroll
        for (int f = 0; f < 4; ++f)
            acc[f] = __builtin_amdgcn_wmma_f32_16x16x32_bf16(
                false, A.v, false, B[f].v, (short)0, acc[f], false, false);
        sched_hint_ds_then_wmma();

        if (more) stregs(buf ^ 1);
        __syncthreads();
        buf ^= 1;
    }

    const int crow = m0 + wm * 16 + khalf * 8;
    const int ccol = n0 + wn * 64 + (lane & 15);
    #pragma unroll
    for (int f = 0; f < 4; ++f)
        #pragma unroll
        for (int r = 0; r < 8; ++r)
            soft[(size_t)(crow + r) * D_MODEL + ccol + f * 16] = acc[f][r];
}

// ---------------------------------------------------------------- residual + LayerNorm
__global__ __launch_bounds__(256) void k_ln(const float* __restrict__ enc,
        const float* __restrict__ soft, const float* __restrict__ gamma,
        const float* __restrict__ beta, float* __restrict__ out)
{
    int row = blockIdx.x, t = threadIdx.x;
    int lane = t & 31, wid = t >> 5;
    size_t base = (size_t)row * D_MODEL;
    float x[4], s = 0.f, s2 = 0.f;
    #pragma unroll
    for (int j = 0; j < 4; ++j) {
        int i = j * 256 + t;
        x[j] = enc[base + i] + soft[base + i];
        s += x[j]; s2 += x[j] * x[j];
    }
    #pragma unroll
    for (int off = 16; off; off >>= 1) {
        s  += __shfl_xor(s, off, 32);
        s2 += __shfl_xor(s2, off, 32);
    }
    __shared__ float rs[8], rs2[8];
    if (lane == 0) { rs[wid] = s; rs2[wid] = s2; }
    __syncthreads();
    float tot = 0.f, tot2 = 0.f;
    #pragma unroll
    for (int w = 0; w < 8; ++w) { tot += rs[w]; tot2 += rs2[w]; }
    float mean = tot * (1.0f / D_MODEL);
    float var  = tot2 * (1.0f / D_MODEL) - mean * mean;
    float rstd = rsqrtf(var + 1e-5f);
    #pragma unroll
    for (int j = 0; j < 4; ++j) {
        int i = j * 256 + t;
        out[base + i] = (x[j] - mean) * rstd * gamma[i] + beta[i];
    }
}

extern "C" void kernel_launch(void* const* d_in, const int* in_sizes, int n_in,
                              void* d_out, int out_size, void* d_ws, size_t ws_size,
                              hipStream_t stream)
{
    const float* enc   = (const float*)d_in[0];
    const float* projw = (const float*)d_in[1];
    const float* projb = (const float*)d_in[2];
    const float* emb   = (const float*)d_in[3];
    const float* gamma = (const float*)d_in[4];
    const float* beta  = (const float*)d_in[5];
    float* out = (float*)d_out;

    // workspace layout (~437 MB): bf16 logits/probs, softmax partials, row stats, soft_emb
    char* ws = (char*)d_ws;
    unsigned short* logits = (unsigned short*)ws;
    size_t off = (size_t)ROWS * VP * sizeof(unsigned short);
    float2* part = (float2*)(ws + off); off += (size_t)ROWS * NBLK * sizeof(float2);
    float*  rowM = (float*)(ws + off);  off += (size_t)ROWS * sizeof(float);
    float*  rowI = (float*)(ws + off);  off += (size_t)ROWS * sizeof(float);
    float*  soft = (float*)(ws + off);  off += (size_t)ROWS * D_MODEL * sizeof(float);
    (void)ws_size; (void)in_sizes; (void)n_in; (void)out_size;

    dim3 g1(NBLK, ROWS / BM);
    k_gemm1<<<g1, 256, 0, stream>>>(enc, projw, projb, logits, part);
    k_softmax_reduce<<<ROWS, 256, 0, stream>>>(part, rowM, rowI);
    dim3 g3((VP / 2 + 255) / 256, ROWS);
    k_softmaxify<<<g3, 256, 0, stream>>>((unsigned int*)logits, rowM, rowI);
    dim3 g4(D_MODEL / BN, ROWS / BM);
    k_gemm2<<<g4, 256, 0, stream>>>((const unsigned int*)logits, emb, soft);
    k_ln<<<ROWS, 256, 0, stream>>>(enc, soft, gamma, beta, out);
}
